// ROIHead_8160437862426
// MI455X (gfx1250) — compile-verified, matrix operand order checked by reading, not run on hardware
//
#include <hip/hip_runtime.h>
#include <hip/hip_bf16.h>
#include <stdint.h>

// ---------- types for WMMA / TDM ----------
typedef __attribute__((ext_vector_type(16))) __bf16 v16bf;
typedef __attribute__((ext_vector_type(8)))  float  v8f;
typedef __attribute__((ext_vector_type(4)))  unsigned v4u;
typedef __attribute__((ext_vector_type(8)))  int    v8i;
typedef __attribute__((ext_vector_type(4)))  int    v4i;

union AFrag { uint4 q[2]; v16bf v; };
union CFrag { v8f v; float f[8]; };

#if defined(__has_builtin)
#if __has_builtin(__builtin_amdgcn_tensor_load_to_lds) && __has_builtin(__builtin_amdgcn_s_wait_tensorcnt)
#define USE_TDM 1
#endif
#endif

// ---------- helpers ----------
__device__ __forceinline__ unsigned short f2bf(float f) {
    unsigned u = __float_as_uint(f);
    unsigned r = u + 0x7FFFu + ((u >> 16) & 1u);   // round-to-nearest-even
    return (unsigned short)(r >> 16);
}
__device__ __forceinline__ unsigned pack2(float a, float b) {
    return (unsigned)f2bf(a) | ((unsigned)f2bf(b) << 16);
}
__device__ __forceinline__ uint4 pack8(float4 a, float4 b) {
    return make_uint4(pack2(a.x, a.y), pack2(a.z, a.w), pack2(b.x, b.y), pack2(b.z, b.w));
}

#define LSTR 40            // LDS row stride in ushorts (64B data + 16B pad = 80B)
#define BUFBYTES (128 * LSTR * 2)

#ifdef USE_TDM
// TDM: DMA one 128-row x 32-col bf16 tile (row stride K elements) into LDS,
// inserting 16B of padding after every 64B row (matches LSTR=40).
__device__ __forceinline__ void tdm_load_tileA(const unsigned short* gsrc,
                                               unsigned ldsOff, int K) {
    const unsigned long long ga = (unsigned long long)(size_t)gsrc;
    v4u g0;
    g0[0] = 1u;                                            // D# count=1, user mode
    g0[1] = ldsOff;                                        // lds_addr (bytes)
    g0[2] = (unsigned)(ga & 0xFFFFFFFFull);                // global_addr[31:0]
    g0[3] = (unsigned)((ga >> 32) & 0x01FFFFFFull)         // global_addr[56:32]
          | 0x80000000u;                                   // type=2 ("image")
    v8i g1;
    g1[0] = (int)((1u << 16)      // data_size = 2 bytes
          |       (1u << 20)      // pad_enable
          |       (3u << 22)      // pad_interval: 16 DWORDs (64B) between pads
          |       (3u << 25));    // pad_amount:   4 DWORDs (16B)
    g1[1] = (int)0xFFFF0000u;                 // tensor_dim0 = 0x7FFFFFFF (lo16 @[31:16])
    g1[2] = (int)0xFFFF7FFFu;                 // dim0 hi16 | tensor_dim1 lo16
    g1[3] = (int)(0x7FFFu | (32u << 16));     // dim1 hi16 | tile_dim0 = 32 elems
    g1[4] = 128;                              // tile_dim1 = 128 rows, tile_dim2 = 0
    g1[5] = K;                                // tensor_dim0_stride (elements), lo32
    g1[6] = 0;
    g1[7] = 0;
    v4i z4 = (v4i){0, 0, 0, 0};
#if __clang_major__ >= 23
    v8i z8 = (v8i){0, 0, 0, 0, 0, 0, 0, 0};
    __builtin_amdgcn_tensor_load_to_lds(g0, g1, z4, z4, z8, 0);
#else
    __builtin_amdgcn_tensor_load_to_lds(g0, g1, z4, z4, 0);
#endif
}
#endif

// =====================================================================
// Kernel 1: ROI max-pool (torchvision roi_pool semantics), output bf16
// pooled layout: [1024 (padded N), 25088] with k = c*49 + ph*7 + pw
// =====================================================================
__global__ __launch_bounds__(256)
void roi_pool_kernel(const float* __restrict__ feat, const float* __restrict__ rois,
                     const int* __restrict__ img_h, unsigned short* __restrict__ pooled,
                     int nroi) {
    const int m = blockIdx.x;
    unsigned short* out = pooled + (size_t)m * 25088;
    if (m >= nroi) {                      // zero pad rows so the GEMM needs no bounds checks
        for (int i = threadIdx.x; i < 25088; i += 256) out[i] = 0;
        return;
    }
    __shared__ int s_h0[7], s_h1[7], s_w0[7], s_w1[7];
    if (threadIdx.x < 7) {
        const float scale = exp2f(rintf(log2f(50.0f / (float)img_h[0])));
        const float x1 = rois[m * 4 + 0], y1 = rois[m * 4 + 1];
        const float x2 = rois[m * 4 + 2], y2 = rois[m * 4 + 3];
        const float sw = rintf(x1 * scale), sh = rintf(y1 * scale);
        const float ew = rintf(x2 * scale), eh = rintf(y2 * scale);
        const float rw = fmaxf(ew - sw + 1.0f, 1.0f);
        const float rh = fmaxf(eh - sh + 1.0f, 1.0f);
        const float p = (float)threadIdx.x;
        s_h0[threadIdx.x] = (int)fminf(fmaxf(floorf(p * rh / 7.0f) + sh, 0.0f), 50.0f);
        s_h1[threadIdx.x] = (int)fminf(fmaxf(ceilf((p + 1.0f) * rh / 7.0f) + sh, 0.0f), 50.0f);
        s_w0[threadIdx.x] = (int)fminf(fmaxf(floorf(p * rw / 7.0f) + sw, 0.0f), 50.0f);
        s_w1[threadIdx.x] = (int)fminf(fmaxf(ceilf((p + 1.0f) * rw / 7.0f) + sw, 0.0f), 50.0f);
    }
    __syncthreads();
    for (int idx = threadIdx.x; idx < 512 * 49; idx += 256) {
        const int c  = idx / 49;
        const int b  = idx - c * 49;
        const int ph = b / 7, pw = b - ph * 7;
        const int h0 = s_h0[ph], h1 = s_h1[ph];
        const int w0 = s_w0[pw], w1 = s_w1[pw];
        const float* fc = feat + (size_t)c * 2500;
        float mx = -1e30f;
        for (int ih = h0; ih < h1; ++ih)
            for (int iw = w0; iw < w1; ++iw)
                mx = fmaxf(mx, fc[ih * 50 + iw]);
        const bool valid = (h1 > h0) && (w1 > w0);
        out[idx] = f2bf(valid ? mx : 0.0f);
    }
}

// =====================================================================
// Kernel 2: assemble merged head weights [128 x 1024] f32 (+bias[128])
// rows 0..20 = w_cls, 21..104 = w_box (deltas row 21 + 4k + c), rest 0
// =====================================================================
__global__ __launch_bounds__(256)
void assemble_head_kernel(const float* __restrict__ w_cls, const float* __restrict__ b_cls,
                          const float* __restrict__ w_box, const float* __restrict__ b_box,
                          float* __restrict__ whead, float* __restrict__ bhead) {
    const int i = blockIdx.x * 256 + threadIdx.x;
    if (i < 128 * 1024) {
        const int r = i >> 10, c = i & 1023;
        float v = 0.0f;
        if (r < 21)       v = w_cls[r * 1024 + c];
        else if (r < 105) v = w_box[(r - 21) * 1024 + c];
        whead[i] = v;
    }
    if (i < 128) {
        float v = 0.0f;
        if (i < 21)       v = b_cls[i];
        else if (i < 105) v = b_box[i - 21];
        bhead[i] = v;
    }
}

// =====================================================================
// Kernel 3: WMMA GEMM  C[M x N] = A[M x K](bf16) * B[N x K](f32)^T + bias
// M = 1024 (padded), N multiple of 128, K multiple of 32.
// Block = 256 thr = 8 waves; block tile 128x128; wave tile 64x32; BK=32.
// A tile staged by the Tensor Data Mover (TDM) when available; B is
// converted f32->bf16 through registers while staging into LDS.
// =====================================================================
__global__ __launch_bounds__(256)
void gemm_wmma_bf16(const unsigned short* __restrict__ A, const float* __restrict__ B,
                    const float* __restrict__ bias, void* __restrict__ Cout,
                    int K, int ldc, int relu, int out_bf16) {
    __shared__ __align__(16) unsigned short sA[2][128 * LSTR];
    __shared__ __align__(16) unsigned short sB[2][128 * LSTR];

    const int tid  = threadIdx.x;
    const int wave = tid >> 5;
    const int lane = tid & 31;
    const int half = lane >> 4;
    const int l15  = lane & 15;
    const int wm   = wave >> 2;          // 0..1  -> 64-row slab
    const int wn   = wave & 3;           // 0..3  -> 32-col slab
    const int mBase = blockIdx.y * 128;
    const int nBase = blockIdx.x * 128;

    // cooperative tile loaders: thread -> (row, 16-element half of the 32-wide K tile)
    const int lr = tid >> 1;
    const int lh = tid & 1;
    const float* gB = B + (size_t)(nBase + lr) * K + lh * 16;
    const int lOff = lr * LSTR + lh * 16;
#ifdef USE_TDM
    const unsigned sAoff = (unsigned)(size_t)&sA[0][0];   // LDS byte offset (addr[31:0])
    const unsigned short* gAtile = A + (size_t)mBase * K; // tile base, row stride K
#else
    const unsigned short* gA = A + (size_t)(mBase + lr) * K + lh * 16;
#endif

    v8f acc[4][2];
#pragma unroll
    for (int i = 0; i < 4; ++i)
#pragma unroll
        for (int j = 0; j < 2; ++j) acc[i][j] = (v8f){};

    const int nk = K >> 5;

    // ---- preload tile 0 ----
#ifdef USE_TDM
    if (wave == 0) tdm_load_tileA(gAtile, sAoff, K);
#else
    {
        const uint4* pA = (const uint4*)gA;
        uint4 a0 = pA[0], a1 = pA[1];
        uint4* dA = (uint4*)&sA[0][lOff];
        dA[0] = a0; dA[1] = a1;
    }
#endif
    {
        const float4* pB = (const float4*)gB;
        float4 f0 = pB[0], f1 = pB[1], f2 = pB[2], f3 = pB[3];
        uint4* dB = (uint4*)&sB[0][lOff];
        dB[0] = pack8(f0, f1); dB[1] = pack8(f2, f3);
    }
#ifdef USE_TDM
    if (wave == 0) __builtin_amdgcn_s_wait_tensorcnt((short)0);
#endif
    __syncthreads();

    for (int kt = 0; kt < nk; ++kt) {
        const int cur = kt & 1;
        const int nxt = cur ^ 1;
        const bool pf = (kt + 1) < nk;
        uint4 nb0, nb1;
#ifndef USE_TDM
        uint4 na0, na1;
#endif
        if (pf) {
#ifdef USE_TDM
            if (wave == 0)
                tdm_load_tileA(gAtile + (size_t)(kt + 1) * 32,
                               sAoff + (unsigned)nxt * BUFBYTES, K);
#else
            const uint4* pA = (const uint4*)(gA + (size_t)(kt + 1) * 32);
            na0 = pA[0]; na1 = pA[1];
#endif
            const float4* pB = (const float4*)(gB + (size_t)(kt + 1) * 32);
            float4 f0 = pB[0], f1 = pB[1], f2 = pB[2], f3 = pB[3];
            nb0 = pack8(f0, f1); nb1 = pack8(f2, f3);
        }

        // ---- compute on current buffer ----
        const unsigned short* bufA = sA[cur];
        const unsigned short* bufB = sB[cur];
        AFrag af[4], bf[2];
#pragma unroll
        for (int fm = 0; fm < 4; ++fm) {
            const unsigned short* pr = bufA + (wm * 64 + fm * 16 + l15) * LSTR;
            af[fm].q[0] = *(const uint4*)(pr + half * 8);
            af[fm].q[1] = *(const uint4*)(pr + 16 + half * 8);
        }
#pragma unroll
        for (int fn = 0; fn < 2; ++fn) {
            const unsigned short* pr = bufB + (wn * 32 + fn * 16 + l15) * LSTR;
            bf[fn].q[0] = *(const uint4*)(pr + half * 8);
            bf[fn].q[1] = *(const uint4*)(pr + 16 + half * 8);
        }
#pragma unroll
        for (int fm = 0; fm < 4; ++fm)
#pragma unroll
            for (int fn = 0; fn < 2; ++fn)
                acc[fm][fn] = __builtin_amdgcn_wmma_f32_16x16x32_bf16(
                    false, af[fm].v, false, bf[fn].v, (short)0, acc[fm][fn], false, false);

        // ---- stage next tile ----
        if (pf) {
            uint4* dB = (uint4*)&sB[nxt][lOff];
            dB[0] = nb0; dB[1] = nb1;
#ifndef USE_TDM
            uint4* dA = (uint4*)&sA[nxt][lOff];
            dA[0] = na0; dA[1] = na1;
#else
            if (wave == 0) __builtin_amdgcn_s_wait_tensorcnt((short)0);
#endif
        }
        __syncthreads();
    }

    // ---- epilogue: bias (+ReLU), store f32 or bf16 ----
#pragma unroll
    for (int fn = 0; fn < 2; ++fn) {
        const int col = nBase + wn * 32 + fn * 16 + l15;
        const float bv = bias[col];
#pragma unroll
        for (int fm = 0; fm < 4; ++fm) {
            CFrag cf; cf.v = acc[fm][fn];
            const int row0 = mBase + wm * 64 + fm * 16 + half * 8;
#pragma unroll
            for (int g = 0; g < 8; ++g) {
                float v = cf.f[g] + bv;
                if (relu) v = fmaxf(v, 0.0f);
                const size_t o = (size_t)(row0 + g) * ldc + col;
                if (out_bf16) ((unsigned short*)Cout)[o] = f2bf(v);
                else          ((float*)Cout)[o] = v;
            }
        }
    }
}

// =====================================================================
// Kernel 4: softmax + box decode + clip. One thread per (roi, class 1..20).
// d_out layout: boxes[20000*4] | scores[20000] | labels[20000]
// =====================================================================
__global__ __launch_bounds__(256)
void head_decode_kernel(const float* __restrict__ head_out, const float* __restrict__ rois,
                        const int* __restrict__ img_h_p, const int* __restrict__ img_w_p,
                        float* __restrict__ out, int nroi) {
    const int t = blockIdx.x * 256 + threadIdx.x;
    if (t >= nroi * 20) return;
    const int m = t / 20;
    const int k = (t - m * 20) + 1;                // class 1..20 (drop background)
    const float* s = head_out + (size_t)m * 128;

    float mx = s[0];
#pragma unroll
    for (int j = 1; j < 21; ++j) mx = fmaxf(mx, s[j]);
    float sum = 0.0f;
#pragma unroll
    for (int j = 0; j < 21; ++j) sum += expf(s[j] - mx);
    const float score = expf(s[k] - mx) / sum;

    const float x1 = rois[m * 4 + 0], y1 = rois[m * 4 + 1];
    const float x2 = rois[m * 4 + 2], y2 = rois[m * 4 + 3];
    const float w = x2 - x1, h = y2 - y1;
    const float cx = x1 + 0.5f * w, cy = y1 + 0.5f * h;
    const float d0 = s[21 + 4 * k + 0], d1 = s[21 + 4 * k + 1];
    const float d2 = s[21 + 4 * k + 2], d3 = s[21 + 4 * k + 3];
    const float pcx = d0 * w + cx, pcy = d1 * h + cy;
    const float pw = expf(d2) * w, ph = expf(d3) * h;
    const float W = (float)img_w_p[0], H = (float)img_h_p[0];

    out[(size_t)t * 4 + 0] = fminf(fmaxf(pcx - 0.5f * pw, 0.0f), W);
    out[(size_t)t * 4 + 1] = fminf(fmaxf(pcy - 0.5f * ph, 0.0f), H);
    out[(size_t)t * 4 + 2] = fminf(fmaxf(pcx + 0.5f * pw, 0.0f), W);
    out[(size_t)t * 4 + 3] = fminf(fmaxf(pcy + 0.5f * ph, 0.0f), H);
    out[(size_t)nroi * 80 + t]  = score;
    out[(size_t)nroi * 100 + t] = (float)k;
}

// =====================================================================
// Launch
// =====================================================================
extern "C" void kernel_launch(void* const* d_in, const int* in_sizes, int n_in,
                              void* d_out, int out_size, void* d_ws, size_t ws_size,
                              hipStream_t stream) {
    (void)n_in; (void)out_size; (void)ws_size;
    const float* feat  = (const float*)d_in[0];
    const float* rois  = (const float*)d_in[1];
    const float* w6    = (const float*)d_in[2];
    const float* b6    = (const float*)d_in[3];
    const float* w7    = (const float*)d_in[4];
    const float* b7    = (const float*)d_in[5];
    const float* w_cls = (const float*)d_in[6];
    const float* b_cls = (const float*)d_in[7];
    const float* w_box = (const float*)d_in[8];
    const float* b_box = (const float*)d_in[9];
    const int*   img_h = (const int*)d_in[10];
    const int*   img_w = (const int*)d_in[11];
    const int nroi = in_sizes[1] / 4;              // 1000

    // workspace carve-out (256B aligned)
    char* ws = (char*)d_ws;
    size_t off = 0;
    auto take = [&](size_t bytes) -> char* {
        char* p = ws + off;
        off = (off + bytes + 255) & ~(size_t)255;
        return p;
    };
    unsigned short* pooled   = (unsigned short*)take(1024ull * 25088 * 2); // 51.4 MB bf16
    unsigned short* x6       = (unsigned short*)take(1024ull * 1024 * 2);
    unsigned short* x7       = (unsigned short*)take(1024ull * 1024 * 2);
    float*          whead    = (float*)take(128ull * 1024 * 4);
    float*          bhead    = (float*)take(128ull * 4);
    float*          head_out = (float*)take(1024ull * 128 * 4);

    roi_pool_kernel<<<1024, 256, 0, stream>>>(feat, rois, img_h, pooled, nroi);
    assemble_head_kernel<<<(128 * 1024 + 255) / 256, 256, 0, stream>>>(
        w_cls, b_cls, w_box, b_box, whead, bhead);

    // FC6: [1024 x 25088] * w6^T -> x6 bf16 (bias+ReLU)
    gemm_wmma_bf16<<<dim3(8, 8), 256, 0, stream>>>(pooled, w6, b6, x6, 25088, 1024, 1, 1);
    // FC7: [1024 x 1024] * w7^T -> x7 bf16 (bias+ReLU)
    gemm_wmma_bf16<<<dim3(8, 8), 256, 0, stream>>>(x6, w7, b7, x7, 1024, 1024, 1, 1);
    // heads: [1024 x 1024] * whead^T -> head_out f32 [1024 x 128]
    gemm_wmma_bf16<<<dim3(1, 8), 256, 0, stream>>>(x7, whead, bhead, head_out, 1024, 128, 0, 0);

    head_decode_kernel<<<(nroi * 20 + 255) / 256, 256, 0, stream>>>(
        head_out, rois, img_h, img_w, (float*)d_out, nroi);
}
